// EllipticGNN_35115652612243
// MI455X (gfx1250) — matrix operand order repaired and checked by reading, not compiled, not run on hardware
//
#include <hip/hip_runtime.h>
#include <hip/hip_bf16.h>
#include <stdint.h>

// ---------------- problem constants (match reference) ----------------
#define NN 80000   // nodes
#define EEc 640000 // edges
#define ENc 720000 // edges + self loops
#define FNc 165    // node feat
#define FNP 192    // node feat padded to K-chunk multiple
#define FEc 16     // edge feat
#define DD 128     // hidden
#define HHc 4      // heads
#define CCc 32     // per-head channels
#define EPc 32     // edge proj dim
#define BBc 64     // graphs
#define LLc 3      // layers
#define DH 64
#define DQ 32
#define LPAD 8     // LDS row pad (bf16 elems) to break bank conflicts

typedef __bf16 bf16;
typedef bf16  v16bf __attribute__((ext_vector_type(16)));
typedef bf16  v8bf  __attribute__((ext_vector_type(8)));
typedef bf16  v4bf  __attribute__((ext_vector_type(4)));
typedef float v8f   __attribute__((ext_vector_type(8)));

// ---------------- small device helpers ----------------
__device__ __forceinline__ bf16 tobf(float f) { return (bf16)f; }

__device__ __forceinline__ float wred32(float v) {
#pragma unroll
  for (int off = 16; off > 0; off >>= 1) v += __shfl_xor(v, off, 32);
  return v;
}

// order-preserving float<->uint encoding for atomicMax-based segment_max
__device__ __forceinline__ unsigned fenc(float f) {
  unsigned u = __float_as_uint(f);
  return (u & 0x80000000u) ? ~u : (u | 0x80000000u);
}
__device__ __forceinline__ float fdec(unsigned k) {
  unsigned u = (k & 0x80000000u) ? (k & 0x7fffffffu) : ~k;
  return __uint_as_float(u);
}

__device__ __forceinline__ v16bf frag_from2(v8bf lo, v8bf hi) {
  return __builtin_shufflevector(lo, hi, 0, 1, 2, 3, 4, 5, 6, 7, 8, 9, 10, 11, 12, 13, 14, 15);
}

// ---------------- WMMA fragment loaders (ISA 7.12.2 layouts) ----------------
// A 16x32 bf16 from LDS tile (row stride = Kp+LPAD): lanes 0-15 row M=lane,
// elems e<8 -> K=kc+e, e>=8 -> K=kc+e+8; lanes 16-31 same rows, +8 K offset.
// Two contiguous 8-elem runs -> two ds_load_b128 (row pad breaks bank conflicts).
__device__ __forceinline__ v16bf lds_a_frag(const bf16* At, int stride, int rowBase, int kc) {
  int lane = threadIdx.x & 31;
  const bf16* p = At + (rowBase + (lane & 15)) * stride + kc + ((lane >> 4) << 3);
  v8bf lo = *(const v8bf*)p;
  v8bf hi = *(const v8bf*)(p + 16);
  return frag_from2(lo, hi);
}
// B 32x16 bf16 from TRANSPOSED weights Wt[N][Kp]: col fixed, elems e<8 -> K=kc+e,
// e>=8 -> K=kc+e+8; lanes 16-31 +8 K offset (matches A) -> two global_load_b128.
__device__ __forceinline__ v16bf load_bt(const bf16* Wt, int Kp, int kc, int colBase) {
  int lane = threadIdx.x & 31;
  const bf16* p = Wt + (size_t)(colBase + (lane & 15)) * Kp + kc + ((lane >> 4) << 3);
  v8bf lo = *(const v8bf*)p;
  v8bf hi = *(const v8bf*)(p + 16);
  return frag_from2(lo, hi);
}
__device__ __forceinline__ v8f wmma_bf16(v16bf a, v16bf b, v8f c) {
  return __builtin_amdgcn_wmma_f32_16x16x32_bf16(false, a, false, b, (short)0, c, false, false);
}

// ---------------- utility kernels ----------------
__global__ void fill_f32(float* p, float v, long n) {
  long i = (long)blockIdx.x * blockDim.x + threadIdx.x;
  if (i < n) p[i] = v;
}
__global__ void fill_u32(unsigned* p, unsigned v, long n) {
  long i = (long)blockIdx.x * blockDim.x + threadIdx.x;
  if (i < n) p[i] = v;
}
// W [K,N] f32 row-major  ->  Wt [N,Kp] bf16 (zero-padded K)
__global__ void transpose_w_bf16(const float* __restrict__ W, bf16* __restrict__ Wt,
                                 int K, int N, int Kp) {
  long i = (long)blockIdx.x * blockDim.x + threadIdx.x;
  if (i >= (long)N * Kp) return;
  int col = (int)(i / Kp), k = (int)(i % Kp);
  Wt[i] = (k < K) ? tobf(W[(size_t)k * N + col]) : tobf(0.f);
}

// ---------------- edge projection: ea = relu(LN(edge_attr @ ep_W + b)) ----------------
__global__ void edge_proj(const float* __restrict__ edge_attr, const float* __restrict__ W,
                          const float* __restrict__ b, const float* __restrict__ g,
                          const float* __restrict__ beta, const int* __restrict__ dst,
                          bf16* __restrict__ ea_all, float* __restrict__ loop_sum,
                          float* __restrict__ cnt) {
  int e = blockIdx.x * blockDim.x + threadIdx.x;
  if (e >= EEc) return;
  float in[FEc];
#pragma unroll
  for (int k = 0; k < FEc; ++k) in[k] = edge_attr[(size_t)e * FEc + k];
  float o[EPc]; float s = 0.f, s2 = 0.f;
#pragma unroll 4
  for (int j = 0; j < EPc; ++j) {
    float a = b[j];
#pragma unroll
    for (int k = 0; k < FEc; ++k) a += in[k] * W[k * EPc + j];
    o[j] = a; s += a; s2 += a * a;
  }
  float mean = s * (1.f / EPc);
  float var = s2 * (1.f / EPc) - mean * mean;
  float inv = rsqrtf(var + 1e-5f);
  int d = dst[e];
#pragma unroll 4
  for (int j = 0; j < EPc; ++j) {
    float v = (o[j] - mean) * inv * g[j] + beta[j];
    v = fmaxf(v, 0.f);
    ea_all[(size_t)e * EPc + j] = tobf(v);
    atomicAdd(&loop_sum[(size_t)d * EPc + j], v);
  }
  atomicAdd(&cnt[d], 1.f);
}

__global__ void loop_ea_finalize(const float* __restrict__ loop_sum, const float* __restrict__ cnt,
                                 bf16* __restrict__ ea_all) {
  int n = blockIdx.x * blockDim.x + threadIdx.x;
  if (n >= NN) return;
  float rc = 1.f / fmaxf(cnt[n], 1.f);
#pragma unroll 4
  for (int j = 0; j < EPc; ++j)
    ea_all[(size_t)(EEc + n) * EPc + j] = tobf(loop_sum[(size_t)n * EPc + j] * rc);
}

// ---------------- input projection: h = relu(LN(x @ ip_W + b)) via WMMA ----------------
// 64 rows/block; 8 waves, each owns 16 output cols and 4 M-tiles (24 WMMA/wave, K=192).
__global__ void __launch_bounds__(256) input_proj_wmma(
    const float* __restrict__ x, const bf16* __restrict__ Wt, const float* __restrict__ bias,
    const float* __restrict__ g, const float* __restrict__ beta,
    float* __restrict__ h, bf16* __restrict__ hbf) {
  __shared__ __align__(16) bf16 xt[64 * (FNP + LPAD)];
  __shared__ float tile[64][DD + 4];
  int wave = threadIdx.x >> 5, lane = threadIdx.x & 31;
  int rowBase = blockIdx.x * 64, colBase = wave * 16;
  // stage x tile to LDS as padded bf16
  for (int i = threadIdx.x; i < 64 * FNP; i += 256) {
    int r = i / FNP, k = i % FNP;
    xt[r * (FNP + LPAD) + k] = (k < FNc) ? tobf(x[(size_t)(rowBase + r) * FNc + k]) : tobf(0.f);
  }
  __syncthreads();
  v8f acc[4] = {};
#pragma unroll
  for (int kc = 0; kc < FNP; kc += 32) {
    v16bf b = load_bt(Wt, FNP, kc, colBase);
#pragma unroll
    for (int m = 0; m < 4; ++m) {
      v16bf a = lds_a_frag(xt, FNP + LPAD, m * 16, kc);
      acc[m] = wmma_bf16(a, b, acc[m]);
    }
  }
  int col = colBase + (lane & 15);
  int mOff = (lane >> 4) << 3;
  float bc = bias[col];
#pragma unroll
  for (int m = 0; m < 4; ++m)
#pragma unroll
    for (int v = 0; v < 8; ++v) tile[m * 16 + mOff + v][col] = acc[m][v] + bc;
  __syncthreads();
  // fused LayerNorm + ReLU: wave handles 8 rows
#pragma unroll
  for (int rr = 0; rr < 8; ++rr) {
    int r = wave * 8 + rr;
    float vals[4]; float s = 0.f, s2 = 0.f;
#pragma unroll
    for (int i = 0; i < 4; ++i) { float f = tile[r][lane + 32 * i]; vals[i] = f; s += f; s2 += f * f; }
    s = wred32(s); s2 = wred32(s2);
    float mean = s * (1.f / DD), var = s2 * (1.f / DD) - mean * mean;
    float inv = rsqrtf(var + 1e-5f);
    size_t gr = (size_t)(rowBase + r) * DD;
#pragma unroll
    for (int i = 0; i < 4; ++i) {
      int c = lane + 32 * i;
      float o = fmaxf((vals[i] - mean) * inv * g[c] + beta[c], 0.f);
      h[gr + c] = o; hbf[gr + c] = tobf(o);
    }
  }
}

// ---------------- xl / xr: out = h @ W + b (80000x128 @ 128x128) via WMMA ----------------
__global__ void __launch_bounds__(256) gemm_nodes_wmma(
    const bf16* __restrict__ hbf, const bf16* __restrict__ Wt,
    const float* __restrict__ bias, float* __restrict__ out) {
  __shared__ __align__(16) bf16 atile[64 * (DD + LPAD)];
  int wave = threadIdx.x >> 5, lane = threadIdx.x & 31;
  int rowBase = blockIdx.x * 64, colBase = wave * 16;
  // stage 64x128 bf16 A tile (16B vector copies into padded rows)
  for (int i = threadIdx.x; i < 64 * (DD / 8); i += 256) {
    int r = i / (DD / 8), seg = i % (DD / 8);
    *(v8bf*)&atile[r * (DD + LPAD) + seg * 8] =
        *(const v8bf*)&hbf[(size_t)(rowBase + r) * DD + seg * 8];
  }
  __syncthreads();
  v8f acc[4] = {};
#pragma unroll
  for (int kc = 0; kc < DD; kc += 32) {
    v16bf b = load_bt(Wt, DD, kc, colBase);
#pragma unroll
    for (int m = 0; m < 4; ++m) {
      v16bf a = lds_a_frag(atile, DD + LPAD, m * 16, kc);
      acc[m] = wmma_bf16(a, b, acc[m]);
    }
  }
  int col = colBase + (lane & 15);
  int mOff = (lane >> 4) << 3;
  float bc = bias[col];
#pragma unroll
  for (int m = 0; m < 4; ++m)
#pragma unroll
    for (int v = 0; v < 8; ++v)
      out[(size_t)(rowBase + m * 16 + mOff + v) * DD + col] = acc[m][v] + bc;
}

// ---------------- fused edge transform + GATv2 logits ----------------
// ee tile (64 edges x 128) via WMMA into LDS (never hits HBM), then
// logit[e,h] = sum_c lrelu(xl[src]+xr[dst]+ee)*gatt ; atomicMax segment max.
__global__ void __launch_bounds__(256) edge_logit_wmma(
    const bf16* __restrict__ ea_all, const bf16* __restrict__ Wet,
    const float* __restrict__ xl, const float* __restrict__ xr,
    const float* __restrict__ gatt, const int* __restrict__ src, const int* __restrict__ dst,
    float* __restrict__ logit, unsigned* __restrict__ mxenc) {
  __shared__ __align__(16) bf16 eat[64 * (EPc + LPAD)];
  __shared__ float ee[64][DD + 4];
  __shared__ int ssrc[64], sdst[64];
  int wave = threadIdx.x >> 5, lane = threadIdx.x & 31;
  int eBase = blockIdx.x * 64, colBase = wave * 16;
  // stage 64x32 bf16 ea tile: 256 threads = 64 rows x 4 16B segments
  {
    int r = threadIdx.x >> 2, seg = threadIdx.x & 3;
    *(v8bf*)&eat[r * (EPc + LPAD) + seg * 8] =
        *(const v8bf*)&ea_all[(size_t)(eBase + r) * EPc + seg * 8];
  }
  if (threadIdx.x < 64) {
    int e = eBase + threadIdx.x;
    ssrc[threadIdx.x] = (e < EEc) ? src[e] : (e - EEc);
    sdst[threadIdx.x] = (e < EEc) ? dst[e] : (e - EEc);
  }
  __syncthreads();
  v16bf b = load_bt(Wet, EPc, 0, colBase);
  int col = colBase + (lane & 15);
  int mOff = (lane >> 4) << 3;
#pragma unroll
  for (int m = 0; m < 4; ++m) {
    v16bf a = lds_a_frag(eat, EPc + LPAD, m * 16, 0);
    v8f acc = {};
    acc = wmma_bf16(a, b, acc);
#pragma unroll
    for (int v = 0; v < 8; ++v) ee[m * 16 + mOff + v][col] = acc[v];
  }
  __syncthreads();
  // 256 tasks: 64 edges x 4 heads
  int le = threadIdx.x & 63, hh = threadIdx.x >> 6;
  int s = ssrc[le], d = sdst[le];
  const float* pl = xl + (size_t)s * DD + hh * CCc;
  const float* pr = xr + (size_t)d * DD + hh * CCc;
  const float* pa = gatt + hh * CCc;
  float lg = 0.f;
#pragma unroll 8
  for (int c = 0; c < CCc; ++c) {
    float t = pl[c] + pr[c] + ee[le][hh * CCc + c];
    t = (t > 0.f) ? t : 0.2f * t;
    lg += t * pa[c];
  }
  int ge = eBase + le;
  logit[(size_t)ge * HHc + hh] = lg;
  atomicMax(&mxenc[(size_t)d * HHc + hh], fenc(lg));
}

// ---------------- softmax numerator + segment sum ----------------
__global__ void edge_softmax(float* __restrict__ att, const unsigned* __restrict__ mxenc,
                             const int* __restrict__ dst, float* __restrict__ ssum) {
  int e = blockIdx.x * blockDim.x + threadIdx.x;
  if (e >= ENc) return;
  int d = (e < EEc) ? dst[e] : (e - EEc);
#pragma unroll
  for (int hh = 0; hh < HHc; ++hh) {
    float a = __expf(att[(size_t)e * HHc + hh] - fdec(mxenc[(size_t)d * HHc + hh]));
    att[(size_t)e * HHc + hh] = a;
    atomicAdd(&ssum[(size_t)d * HHc + hh], a);
  }
}

// ---------------- scatter: agg[dst] += alpha * xl[src] (2 edges / 256-thr block) ----------------
__global__ void __launch_bounds__(256) edge_aggregate(
    const float* __restrict__ att, const float* __restrict__ ssum, const float* __restrict__ xl,
    const int* __restrict__ src, const int* __restrict__ dst, float* __restrict__ agg) {
  int e = blockIdx.x * 2 + (threadIdx.x >> 7);
  int c = threadIdx.x & 127;
  int s = (e < EEc) ? src[e] : (e - EEc);
  int d = (e < EEc) ? dst[e] : (e - EEc);
  int hh = c >> 5;
  float alpha = att[(size_t)e * HHc + hh] / (ssum[(size_t)d * HHc + hh] + 1e-16f);
  atomicAdd(&agg[(size_t)d * DD + c], alpha * xl[(size_t)s * DD + c]);
}

// ---------------- node update: h += relu(LN(agg + gbias)) (float4 path) ----------------
__global__ void __launch_bounds__(256) node_update(
    float* __restrict__ h, bf16* __restrict__ hbf, const float* __restrict__ agg,
    const float* __restrict__ gbias, const float* __restrict__ g, const float* __restrict__ beta) {
  int wave = threadIdx.x >> 5, lane = threadIdx.x & 31;
  int row = blockIdx.x * 8 + wave;
  size_t base = (size_t)row * DD;
  int c0 = lane * 4;
  float4 a4 = *(const float4*)(agg + base + c0);
  float4 gb4 = *(const float4*)(gbias + c0);
  float vals[4] = {a4.x + gb4.x, a4.y + gb4.y, a4.z + gb4.z, a4.w + gb4.w};
  float s = 0.f, s2 = 0.f;
#pragma unroll
  for (int i = 0; i < 4; ++i) { s += vals[i]; s2 += vals[i] * vals[i]; }
  s = wred32(s); s2 = wred32(s2);
  float mean = s * (1.f / DD), var = s2 * (1.f / DD) - mean * mean;
  float inv = rsqrtf(var + 1e-5f);
  float4 g4 = *(const float4*)(g + c0);
  float4 be4 = *(const float4*)(beta + c0);
  float4 h4 = *(const float4*)(h + base + c0);
  float gg[4] = {g4.x, g4.y, g4.z, g4.w}, bb[4] = {be4.x, be4.y, be4.z, be4.w};
  float hv[4] = {h4.x, h4.y, h4.z, h4.w};
  float nh[4]; v4bf nb;
#pragma unroll
  for (int i = 0; i < 4; ++i) {
    float o = fmaxf((vals[i] - mean) * inv * gg[i] + bb[i], 0.f);
    nh[i] = hv[i] + o; nb[i] = tobf(nh[i]);
  }
  *(float4*)(h + base + c0) = make_float4(nh[0], nh[1], nh[2], nh[3]);
  *(v4bf*)(hbf + base + c0) = nb;
}

// ---------------- attention pooling ----------------
__global__ void __launch_bounds__(256) pool_score(
    const float* __restrict__ h, const float* __restrict__ W1, const float* __restrict__ b1,
    const float* __restrict__ W2, const float* __restrict__ b2, const int* __restrict__ batch,
    float* __restrict__ sc, unsigned* __restrict__ mxs) {
  int wave = threadIdx.x >> 5, lane = threadIdx.x & 31;
  int n = blockIdx.x * 8 + wave;
  const float* hn = h + (size_t)n * DD;
  float acc = 0.f;
  for (int j = lane; j < DH; j += 32) {
    float a = b1[j];
#pragma unroll 8
    for (int k = 0; k < DD; ++k) a += hn[k] * W1[k * DH + j];
    acc += tanhf(a) * W2[j];
  }
  acc = wred32(acc);
  if (lane == 0) {
    float v = acc + b2[0];
    sc[n] = v;
    atomicMax(&mxs[batch[n]], fenc(v));
  }
}

__global__ void __launch_bounds__(256) pool_accum(
    const float* __restrict__ h, const float* __restrict__ sc, const unsigned* __restrict__ mxs,
    const int* __restrict__ batch, float* __restrict__ sw, float* __restrict__ cntn,
    float* __restrict__ sum_h, float* __restrict__ att_num, unsigned* __restrict__ maxenc) {
  int wave = threadIdx.x >> 5, lane = threadIdx.x & 31;
  int n = blockIdx.x * 8 + wave;
  int b = batch[n];
  float aw = __expf(sc[n] - fdec(mxs[b]));
  if (lane == 0) { atomicAdd(&sw[b], aw); atomicAdd(&cntn[b], 1.f); }
#pragma unroll
  for (int i = 0; i < 4; ++i) {
    int c = lane + 32 * i;
    float hv = h[(size_t)n * DD + c];
    atomicAdd(&sum_h[(size_t)b * DD + c], hv);
    atomicAdd(&att_num[(size_t)b * DD + c], aw * hv);
    atomicMax(&maxenc[(size_t)b * DD + c], fenc(hv));
  }
}

__global__ void build_xp(const float* __restrict__ att_num, const float* __restrict__ sw,
                         const float* __restrict__ sum_h, const float* __restrict__ cntn,
                         const unsigned* __restrict__ maxenc, float* __restrict__ xp) {
  int i = blockIdx.x * blockDim.x + threadIdx.x;
  if (i >= BBc * DD) return;
  int b = i / DD, c = i % DD;
  xp[(size_t)b * 3 * DD + c] = att_num[i] / (sw[b] + 1e-8f);
  xp[(size_t)b * 3 * DD + DD + c] = sum_h[i] / fmaxf(cntn[b], 1.f);
  xp[(size_t)b * 3 * DD + 2 * DD + c] = fdec(maxenc[i]);
}

// ---------------- classifier: Linear -> BatchNorm(train) -> ReLU ----------------
__global__ void mlp_bn(const float* __restrict__ in, const float* __restrict__ W,
                       const float* __restrict__ bias, const float* __restrict__ bg,
                       const float* __restrict__ bb, float* __restrict__ out, int K, int Ncols) {
  int j = blockIdx.x * blockDim.x + threadIdx.x;
  if (j >= Ncols) return;
  float v[BBc]; float s = 0.f, s2 = 0.f;
  for (int i = 0; i < BBc; ++i) {
    float a = bias[j];
    for (int k = 0; k < K; ++k) a += in[(size_t)i * K + k] * W[(size_t)k * Ncols + j];
    v[i] = a; s += a; s2 += a * a;
  }
  float mean = s * (1.f / BBc), var = s2 * (1.f / BBc) - mean * mean;
  float inv = rsqrtf(var + 1e-5f);
  for (int i = 0; i < BBc; ++i)
    out[(size_t)i * Ncols + j] = fmaxf((v[i] - mean) * inv * bg[j] + bb[j], 0.f);
}

__global__ void mlp_final(const float* __restrict__ in, const float* __restrict__ W,
                          const float* __restrict__ bias, float* __restrict__ out) {
  int i = threadIdx.x;
  if (i >= BBc * 2) return;
  int b = i >> 1, j = i & 1;
  float a = bias[j];
#pragma unroll
  for (int k = 0; k < DQ; ++k) a += in[b * DQ + k] * W[k * 2 + j];
  out[b * 2 + j] = a;
}

// ---------------- host orchestration ----------------
static inline int gridFor(long n, int b) { return (int)((n + b - 1) / b); }
static inline size_t align256(size_t x) { return (x + 255) & ~(size_t)255; }

extern "C" void kernel_launch(void* const* d_in, const int* in_sizes, int n_in,
                              void* d_out, int out_size, void* d_ws, size_t ws_size,
                              hipStream_t stream) {
  (void)in_sizes; (void)n_in; (void)out_size; (void)ws_size;
  const float* x = (const float*)d_in[0];
  const int* edge_index = (const int*)d_in[1];
  const float* edge_attr = (const float*)d_in[2];
  const int* batch = (const int*)d_in[3];
  const float* ep_W = (const float*)d_in[4];
  const float* ep_b = (const float*)d_in[5];
  const float* ep_g = (const float*)d_in[6];
  const float* ep_beta = (const float*)d_in[7];
  const float* ip_W = (const float*)d_in[8];
  const float* ip_b = (const float*)d_in[9];
  const float* ip_g = (const float*)d_in[10];
  const float* ip_beta = (const float*)d_in[11];
  const float* gWl = (const float*)d_in[12];
  const float* gbl = (const float*)d_in[13];
  const float* gWr = (const float*)d_in[14];
  const float* gbr = (const float*)d_in[15];
  const float* gWe = (const float*)d_in[16];
  const float* gatt = (const float*)d_in[17];
  const float* gbias = (const float*)d_in[18];
  const float* ln_g = (const float*)d_in[19];
  const float* ln_b = (const float*)d_in[20];
  const float* ap_W1 = (const float*)d_in[21];
  const float* ap_b1 = (const float*)d_in[22];
  const float* ap_W2 = (const float*)d_in[23];
  const float* ap_b2 = (const float*)d_in[24];
  const float* cW1 = (const float*)d_in[25];
  const float* cb1 = (const float*)d_in[26];
  const float* bn1_g = (const float*)d_in[27];
  const float* bn1_b = (const float*)d_in[28];
  const float* cW2 = (const float*)d_in[29];
  const float* cb2 = (const float*)d_in[30];
  const float* bn2_g = (const float*)d_in[31];
  const float* bn2_b = (const float*)d_in[32];
  const float* cW3 = (const float*)d_in[33];
  const float* cb3 = (const float*)d_in[34];
  const float* bn3_g = (const float*)d_in[35];
  const float* bn3_b = (const float*)d_in[36];
  const float* cW4 = (const float*)d_in[37];
  const float* cb4 = (const float*)d_in[38];
  const int* src = edge_index;
  const int* dst = edge_index + EEc;
  float* outp = (float*)d_out;

  char* ws = (char*)d_ws;
  size_t off = 0;
  auto alloc = [&](size_t bytes) -> void* { void* p = ws + off; off = align256(off + bytes); return p; };
  bf16* ea_all    = (bf16*)alloc((size_t)ENc * EPc * 2);
  float* loop_sum = (float*)alloc((size_t)NN * EPc * 4);
  float* cnt      = (float*)alloc((size_t)NN * 4);
  float* h        = (float*)alloc((size_t)NN * DD * 4);
  bf16* hbf       = (bf16*)alloc((size_t)NN * DD * 2);
  float* xl       = (float*)alloc((size_t)NN * DD * 4);
  float* xr       = (float*)alloc((size_t)NN * DD * 4);
  float* agg      = (float*)alloc((size_t)NN * DD * 4);
  float* att      = (float*)alloc((size_t)ENc * HHc * 4);
  unsigned* mx    = (unsigned*)alloc((size_t)NN * HHc * 4);
  float* ssum     = (float*)alloc((size_t)NN * HHc * 4);
  bf16* ipWt      = (bf16*)alloc((size_t)DD * FNP * 2);
  bf16* gWlt      = (bf16*)alloc((size_t)LLc * DD * DD * 2);
  bf16* gWrt      = (bf16*)alloc((size_t)LLc * DD * DD * 2);
  bf16* gWet      = (bf16*)alloc((size_t)LLc * DD * EPc * 2);
  float* sc       = (float*)alloc((size_t)NN * 4);
  unsigned* mxs   = (unsigned*)alloc((size_t)BBc * 4);
  float* sw       = (float*)alloc((size_t)BBc * 4);
  float* cntn     = (float*)alloc((size_t)BBc * 4);
  float* sum_h    = (float*)alloc((size_t)BBc * DD * 4);
  float* att_num  = (float*)alloc((size_t)BBc * DD * 4);
  unsigned* maxenc = (unsigned*)alloc((size_t)BBc * DD * 4);
  float* xp       = (float*)alloc((size_t)BBc * 3 * DD * 4);
  float* z1       = (float*)alloc((size_t)BBc * DD * 4);
  float* z2       = (float*)alloc((size_t)BBc * DH * 4);
  float* z3       = (float*)alloc((size_t)BBc * DQ * 4);

  // weight transpose+convert to bf16 [N][Kp]
  transpose_w_bf16<<<gridFor((long)DD * FNP, 256), 256, 0, stream>>>(ip_W, ipWt, FNc, DD, FNP);
  for (int l = 0; l < LLc; ++l) {
    transpose_w_bf16<<<gridFor((long)DD * DD, 256), 256, 0, stream>>>(
        gWl + (size_t)l * DD * DD, gWlt + (size_t)l * DD * DD, DD, DD, DD);
    transpose_w_bf16<<<gridFor((long)DD * DD, 256), 256, 0, stream>>>(
        gWr + (size_t)l * DD * DD, gWrt + (size_t)l * DD * DD, DD, DD, DD);
    transpose_w_bf16<<<gridFor((long)DD * EPc, 256), 256, 0, stream>>>(
        gWe + (size_t)l * EPc * DD, gWet + (size_t)l * DD * EPc, EPc, DD, EPc);
  }

  // edge projection + self-loop mean edge attr
  fill_f32<<<gridFor((long)NN * EPc, 256), 256, 0, stream>>>(loop_sum, 0.f, (long)NN * EPc);
  fill_f32<<<gridFor(NN, 256), 256, 0, stream>>>(cnt, 0.f, NN);
  edge_proj<<<gridFor(EEc, 256), 256, 0, stream>>>(edge_attr, ep_W, ep_b, ep_g, ep_beta, dst,
                                                   ea_all, loop_sum, cnt);
  loop_ea_finalize<<<gridFor(NN, 256), 256, 0, stream>>>(loop_sum, cnt, ea_all);

  // input projection (WMMA + fused LN/ReLU)
  input_proj_wmma<<<NN / 64, 256, 0, stream>>>(x, ipWt, ip_b, ip_g, ip_beta, h, hbf);

  // GATv2 layers
  for (int l = 0; l < LLc; ++l) {
    gemm_nodes_wmma<<<NN / 64, 256, 0, stream>>>(hbf, gWlt + (size_t)l * DD * DD, gbl + l * DD, xl);
    gemm_nodes_wmma<<<NN / 64, 256, 0, stream>>>(hbf, gWrt + (size_t)l * DD * DD, gbr + l * DD, xr);
    fill_u32<<<gridFor((long)NN * HHc, 256), 256, 0, stream>>>(mx, 0u, (long)NN * HHc);
    fill_f32<<<gridFor((long)NN * HHc, 256), 256, 0, stream>>>(ssum, 0.f, (long)NN * HHc);
    fill_f32<<<gridFor((long)NN * DD, 256), 256, 0, stream>>>(agg, 0.f, (long)NN * DD);
    edge_logit_wmma<<<ENc / 64, 256, 0, stream>>>(ea_all, gWet + (size_t)l * DD * EPc, xl, xr,
                                                  gatt + l * HHc * CCc, src, dst, att, mx);
    edge_softmax<<<gridFor(ENc, 256), 256, 0, stream>>>(att, mx, dst, ssum);
    edge_aggregate<<<ENc / 2, 256, 0, stream>>>(att, ssum, xl, src, dst, agg);
    node_update<<<NN / 8, 256, 0, stream>>>(h, hbf, agg, gbias + l * DD, ln_g + l * DD, ln_b + l * DD);
  }

  // pooling
  fill_u32<<<1, 256, 0, stream>>>(mxs, 0u, BBc);
  fill_f32<<<1, 256, 0, stream>>>(sw, 0.f, BBc);
  fill_f32<<<1, 256, 0, stream>>>(cntn, 0.f, BBc);
  fill_f32<<<gridFor((long)BBc * DD, 256), 256, 0, stream>>>(sum_h, 0.f, (long)BBc * DD);
  fill_f32<<<gridFor((long)BBc * DD, 256), 256, 0, stream>>>(att_num, 0.f, (long)BBc * DD);
  fill_u32<<<gridFor((long)BBc * DD, 256), 256, 0, stream>>>(maxenc, 0u, (long)BBc * DD);
  pool_score<<<NN / 8, 256, 0, stream>>>(h, ap_W1, ap_b1, ap_W2, ap_b2, batch, sc, mxs);
  pool_accum<<<NN / 8, 256, 0, stream>>>(h, sc, mxs, batch, sw, cntn, sum_h, att_num, maxenc);
  build_xp<<<gridFor((long)BBc * DD, 256), 256, 0, stream>>>(att_num, sw, sum_h, cntn, maxenc, xp);

  // classifier head
  mlp_bn<<<1, 128, 0, stream>>>(xp, cW1, cb1, bn1_g, bn1_b, z1, 3 * DD, DD);
  mlp_bn<<<1, 64, 0, stream>>>(z1, cW2, cb2, bn2_g, bn2_b, z2, DD, DH);
  mlp_bn<<<1, 32, 0, stream>>>(z2, cW3, cb3, bn3_g, bn3_b, z3, DH, DQ);
  mlp_final<<<1, 128, 0, stream>>>(z3, cW4, cb4, outp);
}
